// MoELayer_10015863734803
// MI455X (gfx1250) — compile-verified, matrix operand order checked by reading, not run on hardware
//
#include <hip/hip_runtime.h>
#include <hip/hip_bf16.h>
#include <stdint.h>

#define NUM_EXPERTS 8
#define TOP_K 2
#define D_MODEL 1024
#define D_FF 4096
#define T_TOKENS 4096

// LDS row strides (elements). stride*2 bytes = 4 dwords (mod 64 banks) per row,
// so lanes 0..15 (rows 0..15) cover all 64 banks on ds_load_b128: conflict-free.
#define XS_STRIDE 1032
#define HS_STRIDE 4104
#define SMEM_BYTES (128 + 16 * XS_STRIDE * 2 + 16 * HS_STRIDE * 2)

// Packed-weight geometry: B-tiles of 32(K) x 16(N) bf16, 256 dwords (1KB) per tile,
// stored so lane l reads its 8 B-operand dwords as two contiguous uint4.
#define W1_NT (D_FF / 16)     // 256 n-tiles
#define W1_KK (D_MODEL / 32)  // 32 k-tiles
#define W2_NT (D_MODEL / 16)  // 64 n-tiles
#define W2_KK (D_FF / 32)     // 128 k-tiles
#define W1_TILES (NUM_EXPERTS * W1_NT * W1_KK)  // 65536
#define W2_TILES (NUM_EXPERTS * W2_NT * W2_KK)  // 65536

typedef __attribute__((ext_vector_type(16))) __bf16 v16bf;
typedef __attribute__((ext_vector_type(8))) float v8f;

union ABu {
  unsigned u[8];
  v16bf v;
};

__device__ __forceinline__ unsigned short cvt_bf16(float f) {
  unsigned a = __float_as_uint(f);
  a = (a + 0x7FFFu + ((a >> 16) & 1u)) >> 16;  // round-to-nearest-even
  return (unsigned short)a;
}

__device__ __forceinline__ unsigned cvt2_bf16(float lo, float hi) {
  return (unsigned)cvt_bf16(lo) | ((unsigned)cvt_bf16(hi) << 16);
}

// ---------------------------------------------------------------------------
// Kernel 0: zero output accumulator and per-expert routing counters
// ---------------------------------------------------------------------------
__global__ void moe_zero_kernel(float* __restrict__ out, int n, int* __restrict__ counts) {
  int i = blockIdx.x * blockDim.x + threadIdx.x;
  if (i < n) out[i] = 0.0f;
  if (i < NUM_EXPERTS) counts[i] = 0;
}

// ---------------------------------------------------------------------------
// Kernel 1: gating (x @ gate_w), top-2 + softmax, compact per-expert lists.
// One wave32 per token.
// ---------------------------------------------------------------------------
__global__ __launch_bounds__(128) void moe_gate_kernel(
    const float* __restrict__ x, const float* __restrict__ gw,
    int* __restrict__ counts, int* __restrict__ ids, float* __restrict__ wts) {
  const int t = blockIdx.x * 4 + (threadIdx.x >> 5);
  const int lane = threadIdx.x & 31;

  float acc[NUM_EXPERTS];
#pragma unroll
  for (int e = 0; e < NUM_EXPERTS; ++e) acc[e] = 0.0f;

  const float* xr = x + (size_t)t * D_MODEL;
  for (int d = lane; d < D_MODEL; d += 32) {
    const float xv = xr[d];
#pragma unroll
    for (int e = 0; e < NUM_EXPERTS; ++e) acc[e] += xv * gw[d * NUM_EXPERTS + e];
  }
#pragma unroll
  for (int off = 16; off > 0; off >>= 1) {
#pragma unroll
    for (int e = 0; e < NUM_EXPERTS; ++e) acc[e] += __shfl_xor(acc[e], off, 32);
  }

  if (lane == 0) {
    int i1 = 0;
#pragma unroll
    for (int e = 1; e < NUM_EXPERTS; ++e)
      if (acc[e] > acc[i1]) i1 = e;
    int i2 = (i1 == 0) ? 1 : 0;
#pragma unroll
    for (int e = 0; e < NUM_EXPERTS; ++e)
      if (e != i1 && acc[e] > acc[i2]) i2 = e;

    const float e2 = __expf(acc[i2] - acc[i1]);  // exp of max is 1
    const float s = 1.0f / (1.0f + e2);

    int p1 = atomicAdd(&counts[i1], 1);
    ids[i1 * T_TOKENS + p1] = t;
    wts[i1 * T_TOKENS + p1] = s;
    int p2 = atomicAdd(&counts[i2], 1);
    ids[i2 * T_TOKENS + p2] = t;
    wts[i2 * T_TOKENS + p2] = e2 * s;
  }
}

// ---------------------------------------------------------------------------
// Kernel 2: pre-pack fp32 weights -> bf16 in WMMA B-operand tile order.
// One thread produces one lane's 8 dwords of one 32x16 tile (two uint4 stores).
// ---------------------------------------------------------------------------
__global__ __launch_bounds__(256) void moe_pack_kernel(
    const float* __restrict__ w1, const float* __restrict__ w2,
    unsigned* __restrict__ bw1, unsigned* __restrict__ bw2) {
  const int t = blockIdx.x * 256 + threadIdx.x;
  const int lane = t & 31;
  const int tile = t >> 5;
  const int m = lane & 15, g = lane >> 4;
  unsigned o[8];

  if (tile < W1_TILES) {
    const int kk = tile & (W1_KK - 1);
    const int nt = (tile >> 5) & (W1_NT - 1);
    const int e = tile >> 13;
    const float* src = w1 + (size_t)e * D_MODEL * D_FF + nt * 16 + m;
#pragma unroll
    for (int v = 0; v < 8; ++v) {
      const int k = kk * 32 + ((v >= 4) ? 16 : 0) + g * 8 + (v & 3) * 2;
      o[v] = cvt2_bf16(src[(size_t)k * D_FF], src[(size_t)(k + 1) * D_FF]);
    }
    unsigned* dst = bw1 + (size_t)tile * 256 + lane * 8;
    *(uint4*)dst = make_uint4(o[0], o[1], o[2], o[3]);
    *(uint4*)(dst + 4) = make_uint4(o[4], o[5], o[6], o[7]);
  } else {
    const int t2 = tile - W1_TILES;
    const int kk = t2 & (W2_KK - 1);
    const int nt = (t2 >> 7) & (W2_NT - 1);
    const int e = t2 >> 13;
    const float* src = w2 + (size_t)e * D_FF * D_MODEL + nt * 16 + m;
#pragma unroll
    for (int v = 0; v < 8; ++v) {
      const int k = kk * 32 + ((v >= 4) ? 16 : 0) + g * 8 + (v & 3) * 2;
      o[v] = cvt2_bf16(src[(size_t)k * D_MODEL], src[(size_t)(k + 1) * D_MODEL]);
    }
    unsigned* dst = bw2 + (size_t)t2 * 256 + lane * 8;
    *(uint4*)dst = make_uint4(o[0], o[1], o[2], o[3]);
    *(uint4*)(dst + 4) = make_uint4(o[4], o[5], o[6], o[7]);
  }
}

// ---------------------------------------------------------------------------
// Kernel 3: fused expert FFN for a tile of 16 tokens of one expert.
//   h = relu(x @ W1[e])  kept entirely in LDS as bf16 (16 x 4096, 320KB WGP LDS)
//   out += combine_w * (h @ W2[e])  via global_atomic_add_f32
// PACKED: B operands come pre-tiled bf16 (2x global_load_b128 per WMMA).
// ---------------------------------------------------------------------------
template <bool PACKED>
__global__ __launch_bounds__(256) void moe_ffn_kernel(
    const float* __restrict__ x, const float* __restrict__ w1,
    const float* __restrict__ w2, const unsigned* __restrict__ bw1,
    const unsigned* __restrict__ bw2, const int* __restrict__ counts,
    const int* __restrict__ ids, const float* __restrict__ wts,
    float* __restrict__ out) {
  extern __shared__ char smem[];
  int* tok = (int*)smem;                               // 16 ints
  float* twt = (float*)(smem + 64);                    // 16 floats
  unsigned short* xs = (unsigned short*)(smem + 128);  // [16][XS_STRIDE] bf16
  unsigned short* hs = xs + 16 * XS_STRIDE;            // [16][HS_STRIDE] bf16

  const int e = blockIdx.y;
  const int row0 = blockIdx.x * 16;
  const int cnt = counts[e];
  if (row0 >= cnt) return;
  const int rows = min(16, cnt - row0);

  const int tid = threadIdx.x;
  if (tid < 16) {
    const int r = tid;
    const int src = e * T_TOKENS + row0 + (r < rows ? r : 0);
    tok[r] = ids[src];
    twt[r] = (r < rows) ? wts[src] : 0.0f;  // padded rows contribute 0
  }
  __syncthreads();

  // stage gathered activation rows into LDS as bf16
  for (int i = tid; i < 16 * D_MODEL; i += 256) {
    const int r = i >> 10, c = i & (D_MODEL - 1);
    xs[r * XS_STRIDE + c] = cvt_bf16(x[(size_t)tok[r] * D_MODEL + c]);
  }
  __syncthreads();

  const int wave = tid >> 5;
  const int lane = tid & 31;
  const int m = lane & 15;  // A row / B col / D col index
  const int g = lane >> 4;  // lane half-group

  // ----- Phase 1: h[16, D_FF] = relu(x @ W1[e]) -----
  {
    const unsigned short* arow = xs + m * XS_STRIDE;
    for (int nt = wave; nt < W1_NT; nt += 8) {
      const int n0 = nt * 16;
      const unsigned* bt = bw1 + ((size_t)(e * W1_NT + nt) * W1_KK) * 256 + lane * 8;
      const float* bcol = w1 + (size_t)e * D_MODEL * D_FF + n0 + m;
      v8f acc = {0.f, 0.f, 0.f, 0.f, 0.f, 0.f, 0.f, 0.f};
#pragma unroll 4
      for (int kk = 0; kk < W1_KK; ++kk) {
        ABu a, b;
        const uint4 lo = *(const uint4*)(arow + kk * 32 + g * 8);
        const uint4 hi = *(const uint4*)(arow + kk * 32 + 16 + g * 8);
        a.u[0] = lo.x; a.u[1] = lo.y; a.u[2] = lo.z; a.u[3] = lo.w;
        a.u[4] = hi.x; a.u[5] = hi.y; a.u[6] = hi.z; a.u[7] = hi.w;
        if constexpr (PACKED) {
          const uint4 b0 = *(const uint4*)(bt + (size_t)kk * 256);
          const uint4 b1 = *(const uint4*)(bt + (size_t)kk * 256 + 4);
          __builtin_prefetch(bt + (size_t)(kk + 4) * 256, 0, 1);
          b.u[0] = b0.x; b.u[1] = b0.y; b.u[2] = b0.z; b.u[3] = b0.w;
          b.u[4] = b1.x; b.u[5] = b1.y; b.u[6] = b1.z; b.u[7] = b1.w;
        } else {
#pragma unroll
          for (int v = 0; v < 8; ++v) {
            const int kv = kk * 32 + ((v >= 4) ? 16 : 0) + g * 8 + (v & 3) * 2;
            const size_t off = (size_t)kv * D_FF;
            b.u[v] = cvt2_bf16(bcol[off], bcol[off + D_FF]);
          }
        }
        acc = __builtin_amdgcn_wmma_f32_16x16x32_bf16(false, a.v, false, b.v,
                                                      (short)0, acc, false, false);
      }
#pragma unroll
      for (int r = 0; r < 8; ++r) {
        const int M = r + g * 8;
        const float h = acc[r] > 0.0f ? acc[r] : 0.0f;  // relu
        hs[M * HS_STRIDE + n0 + m] = cvt_bf16(h);
      }
    }
  }
  __syncthreads();

  // ----- Phase 2: out[tok] += twt * (h @ W2[e]) -----
  {
    const unsigned short* arow = hs + m * HS_STRIDE;
    for (int nt = wave; nt < W2_NT; nt += 8) {
      const int n0 = nt * 16;
      const unsigned* bt = bw2 + ((size_t)(e * W2_NT + nt) * W2_KK) * 256 + lane * 8;
      const float* bcol = w2 + (size_t)e * D_FF * D_MODEL + n0 + m;
      v8f acc = {0.f, 0.f, 0.f, 0.f, 0.f, 0.f, 0.f, 0.f};
#pragma unroll 4
      for (int kk = 0; kk < W2_KK; ++kk) {
        ABu a, b;
        const uint4 lo = *(const uint4*)(arow + kk * 32 + g * 8);
        const uint4 hi = *(const uint4*)(arow + kk * 32 + 16 + g * 8);
        a.u[0] = lo.x; a.u[1] = lo.y; a.u[2] = lo.z; a.u[3] = lo.w;
        a.u[4] = hi.x; a.u[5] = hi.y; a.u[6] = hi.z; a.u[7] = hi.w;
        if constexpr (PACKED) {
          const uint4 b0 = *(const uint4*)(bt + (size_t)kk * 256);
          const uint4 b1 = *(const uint4*)(bt + (size_t)kk * 256 + 4);
          __builtin_prefetch(bt + (size_t)(kk + 4) * 256, 0, 1);
          b.u[0] = b0.x; b.u[1] = b0.y; b.u[2] = b0.z; b.u[3] = b0.w;
          b.u[4] = b1.x; b.u[5] = b1.y; b.u[6] = b1.z; b.u[7] = b1.w;
        } else {
#pragma unroll
          for (int v = 0; v < 8; ++v) {
            const int kv = kk * 32 + ((v >= 4) ? 16 : 0) + g * 8 + (v & 3) * 2;
            const size_t off = (size_t)kv * D_MODEL;
            b.u[v] = cvt2_bf16(bcol[off], bcol[off + D_MODEL]);
          }
        }
        acc = __builtin_amdgcn_wmma_f32_16x16x32_bf16(false, a.v, false, b.v,
                                                      (short)0, acc, false, false);
      }
#pragma unroll
      for (int r = 0; r < 8; ++r) {
        const int M = r + g * 8;
        unsafeAtomicAdd(&out[(size_t)tok[M] * D_MODEL + n0 + m], acc[r] * twt[M]);
      }
    }
  }
}

// ---------------------------------------------------------------------------
extern "C" void kernel_launch(void* const* d_in, const int* in_sizes, int n_in,
                              void* d_out, int out_size, void* d_ws, size_t ws_size,
                              hipStream_t stream) {
  const float* x  = (const float*)d_in[0];  // [T, D_MODEL]
  const float* gw = (const float*)d_in[1];  // [D_MODEL, E]
  const float* w1 = (const float*)d_in[2];  // [E, D_MODEL, D_FF]
  const float* w2 = (const float*)d_in[3];  // [E, D_FF, D_MODEL]
  float* out = (float*)d_out;               // [T, D_MODEL]

  char* ws = (char*)d_ws;
  int* counts = (int*)ws;                                        // [E]
  int* ids = (int*)(ws + 256);                                   // [E, T]
  float* wts = (float*)(ws + 256 + NUM_EXPERTS * T_TOKENS * 4);  // [E, T]

  const size_t BW1_OFF = (size_t)1 << 20;
  const size_t BW_U32 = (size_t)W1_TILES * 256;  // 16.78M dwords each
  const size_t BW2_OFF = BW1_OFF + BW_U32 * 4;
  const size_t REQUIRED = BW2_OFF + BW_U32 * 4;  // ~129 MB
  unsigned* bw1 = (unsigned*)(ws + BW1_OFF);
  unsigned* bw2 = (unsigned*)(ws + BW2_OFF);
  const bool packed = ws_size >= REQUIRED;

  const int n = T_TOKENS * D_MODEL;
  moe_zero_kernel<<<(n + 255) / 256, 256, 0, stream>>>(out, n, counts);
  moe_gate_kernel<<<T_TOKENS / 4, 128, 0, stream>>>(x, gw, counts, ids, wts);

  dim3 grid(T_TOKENS / 16, NUM_EXPERTS);
  if (packed) {
    const int pack_threads = (W1_TILES + W2_TILES) * 32;
    moe_pack_kernel<<<pack_threads / 256, 256, 0, stream>>>(w1, w2, bw1, bw2);
    moe_ffn_kernel<true><<<grid, 256, SMEM_BYTES, stream>>>(
        x, w1, w2, bw1, bw2, counts, ids, wts, out);
  } else {
    moe_ffn_kernel<false><<<grid, 256, SMEM_BYTES, stream>>>(
        x, w1, w2, bw1, bw2, counts, ids, wts, out);
  }
}
